// TwoLayerNetwork_8469675508310
// MI455X (gfx1250) — matrix operand order repaired and checked by reading, not compiled, not run on hardware
//
#include <hip/hip_runtime.h>
#include <math.h>

typedef __attribute__((ext_vector_type(16))) _Float16 v16h;
typedef __attribute__((ext_vector_type(8)))  float    v8f;

#define T_STEPS      32
#define ROWS_PER_WG  16
#define THREADS      512
#define FPT          25      // features per thread; 32 chunks * 25 = 800 >= 784
#define KCH          25      // K chunks of 32 (800 = 25*32)
#define NFEAT        784
#define TILE_ELEMS   (KCH*32*16)   // 12800 f16 per operand buffer

__global__ __launch_bounds__(THREADS)
void snn_lif_wmma_kernel(const float* __restrict__ x,
                         const float* __restrict__ W,
                         float* __restrict__ out)
{
    // A-operand (spikes) and B-operand (weights) staged in WMMA register layouts.
    __shared__ __align__(32) _Float16 zlds[TILE_ELEMS];
    __shared__ __align__(32) _Float16 wlds[TILE_ELEMS];
    __shared__ float vmaxlds[16 * 16];

    const int tid     = threadIdx.x;
    const int r       = tid & 15;      // row within 16-row tile
    const int cchunk  = tid >> 4;      // 0..31 feature chunk
    const int rowBase = blockIdx.x * ROWS_PER_WG;
    const int row     = rowBase + r;
    const int lane    = tid & 31;

    // ---- one-time LDS init: zero spike buffer, pack W into B-matrix layout ----
    // B layout (32x16 f16): lanes 0-15 -> N=lane, K rem {0..7,16..23}
    //                       lanes 16-31 -> N=lane-16, K rem {8..15,24..31}
    for (int e = tid; e < TILE_ELEMS; e += THREADS) {
        zlds[e] = (_Float16)0.0f;
        int kk = e >> 9;           // /512
        int ln = (e >> 4) & 31;
        int h  = e & 15;
        int n  = ln & 15;
        int k  = kk * 32 + (h & 7) + ((h & 8) ? 16 : 0) + ((ln & 16) ? 8 : 0);
        float wv = (n < 10 && k < NFEAT) ? W[n * NFEAT + k] : 0.0f;
        wlds[e] = (_Float16)wv;
    }

    // ---- load constant input current once; init recurrent state in registers ----
    float xv[FPT], venc[FPT], v0[FPT], i0[FPT];
#pragma unroll
    for (int j = 0; j < FPT; ++j) {
        int f = cchunk * FPT + j;
        xv[j]   = (f < NFEAT) ? x[row * NFEAT + f] : 0.0f;
        venc[j] = 0.0f; v0[j] = 0.0f; i0[j] = 0.0f;
    }

    v8f v_out, i_out, v_max;
#pragma unroll
    for (int e = 0; e < 8; ++e) { v_out[e] = 0.0f; i_out[e] = 0.0f; v_max[e] = -INFINITY; }

    __syncthreads();

#pragma unroll 1
    for (int t = 0; t < T_STEPS; ++t) {
        // -------- elementwise LIF dynamics (registers), emit z0 spikes to LDS --------
        // A layout (16x32 f16): lane<16 holds M=lane, K rem {0..7,16..23};
        //                       lane>=16 holds M=lane-16, K rem {8..15,24..31}.
#pragma unroll
        for (int j = 0; j < FPT; ++j) {
            int f = cchunk * FPT + j;
            if (f < NFEAT) {
                // ConstantCurrentLIFEncoder
                float ve   = venc[j] + 0.1f * (xv[j] - venc[j]);
                float zenc = (ve > 1.0f) ? 1.0f : 0.0f;
                venc[j]    = ve - zenc * ve;
                // LIFCell
                float vdec = v0[j] + 0.1f * (i0[j] - v0[j]);
                float idec = 0.8f * i0[j];
                float z0   = (vdec > 1.0f) ? 1.0f : 0.0f;
                v0[j]      = (1.0f - z0) * vdec;
                i0[j]      = idec + zenc;
                // scatter spike into A-matrix swizzle
                int kk  = f >> 5;
                int rem = f & 31;
                int zl  = r + ((rem & 8) ? 16 : 0);
                int h   = (rem & 7) + ((rem & 16) ? 8 : 0);
                zlds[(kk * 32 + zl) * 16 + h] = (_Float16)z0;
            }
        }
        __syncthreads();

        // -------- wave 0: i_jump = i_out + z0 @ W^T via chained WMMA --------
        if (tid < 32) {
            v8f c = i_out;   // accumulator seeded with i_out
            for (int kk = 0; kk < KCH; ++kk) {
                v16h a = *(const v16h*)&zlds[(kk * 32 + lane) * 16];
                v16h b = *(const v16h*)&wlds[(kk * 32 + lane) * 16];
                c = __builtin_amdgcn_wmma_f32_16x16x32_f16(
                        /*neg_a=*/false, a, /*neg_b=*/false, b,
                        /*c_mod=*/(short)0, c, /*reuse_a=*/false, /*reuse_b=*/false);
            }
            // LILinearCell leaky integrator + running max, on the C-layout registers
#pragma unroll
            for (int e = 0; e < 8; ++e) {
                float ij = c[e];
                float vo = v_out[e] + 0.1f * (ij - v_out[e]);
                v_out[e] = vo;
                i_out[e] = 0.8f * ij;
                v_max[e] = fmaxf(v_max[e], vo);
            }
        }
        __syncthreads();   // protect zlds from next step's writes
    }

    // -------- spill v_max (C layout: VGPR e -> M=e or e+8, lane -> N) --------
    if (tid < 32) {
#pragma unroll
        for (int e = 0; e < 8; ++e) {
            int m = (lane < 16) ? e : (e + 8);
            int n = lane & 15;
            vmaxlds[m * 16 + n] = v_max[e];
        }
    }
    __syncthreads();

    // -------- log_softmax over the 10 valid outputs per row --------
    if (tid < ROWS_PER_WG * 10) {
        int rr = tid / 10, o = tid % 10;
        float m = -INFINITY;
#pragma unroll
        for (int q = 0; q < 10; ++q) m = fmaxf(m, vmaxlds[rr * 16 + q]);
        float s = 0.0f;
#pragma unroll
        for (int q = 0; q < 10; ++q) s += __expf(vmaxlds[rr * 16 + q] - m);
        out[(rowBase + rr) * 10 + o] = (vmaxlds[rr * 16 + o] - m) - __logf(s);
    }
}

extern "C" void kernel_launch(void* const* d_in, const int* in_sizes, int n_in,
                              void* d_out, int out_size, void* d_ws, size_t ws_size,
                              hipStream_t stream) {
    const float* x   = (const float*)d_in[0];   // [B,1,28,28] -> [B,784]
    const float* W   = (const float*)d_in[1];   // [10,784]
    float*       out = (float*)d_out;           // [B,10]
    const int B = in_sizes[0] / NFEAT;          // 4096
    const int grid = (B + ROWS_PER_WG - 1) / ROWS_PER_WG;
    snn_lif_wmma_kernel<<<grid, THREADS, 0, stream>>>(x, W, out);
}
